// TransformerLM_78219944394922
// MI455X (gfx1250) — matrix operand order repaired and checked by reading, not compiled, not run on hardware
//
#include <hip/hip_runtime.h>
#include <cstddef>
#include <cstdint>

// ---------------- model dims ----------------
constexpr int V_SZ  = 32000;
constexpr int D_SZ  = 1024;
constexpr int L_NUM = 6;
constexpr int NH    = 16;
constexpr int FF_SZ = 4096;
constexpr int S_SZ  = 1024;
constexpr int B_SZ  = 2;
constexpr int DK    = 64;                 // D/NH
constexpr int M_TOK = B_SZ * S_SZ;        // 2048 rows in all GEMMs

// ---------------- vector types ----------------
typedef __attribute__((ext_vector_type(16))) __bf16        v16bf;
typedef __attribute__((ext_vector_type(8)))  __bf16        v8bf;
typedef __attribute__((ext_vector_type(8)))  float         v8f;
typedef __attribute__((ext_vector_type(8)))  unsigned short us8;

union FragBF { v16bf v; v8bf h[2]; };

__device__ inline unsigned short f32_to_bf16(float f) {
  unsigned u = __float_as_uint(f);
  u += 0x7fffu + ((u >> 16) & 1u);      // round-to-nearest-even
  return (unsigned short)(u >> 16);
}
__device__ inline __bf16 us_to_bf(unsigned short h) {
  union { unsigned short u; __bf16 b; } c; c.u = h; return c.b;
}

// ---------------- CDNA5 async global->LDS (ASYNCcnt) ----------------
// low 32 bits of a generic pointer to
// __shared__ are the LDS byte address (aperture rules, ISA 10.2).
__device__ inline void async_ld_b128(void* lds, const void* gptr) {
  unsigned           l = (unsigned)(unsigned long long)lds;
  unsigned long long g = (unsigned long long)gptr;
  asm volatile("global_load_async_to_lds_b128 %0, %1, off"
               :: "v"(l), "v"(g) : "memory");
}
__device__ inline void wait_async0() {
  asm volatile("s_wait_asynccnt 0x0" ::: "memory");
}

// ---------------- f32 -> bf16 conversion ----------------
__global__ __launch_bounds__(256) void cvt_bf16_kernel(
    const float* __restrict__ in, unsigned short* __restrict__ out, size_t n) {
  size_t i = (size_t)blockIdx.x * blockDim.x + threadIdx.x;
  if (i < n) out[i] = f32_to_bf16(in[i]);
}

// ---------------- embedding + sinusoidal PE ----------------
__global__ __launch_bounds__(256) void embed_kernel(
    const int* __restrict__ x, const float* __restrict__ emb,
    float* __restrict__ h) {
  const int row = blockIdx.x;                 // 0 .. B*S-1
  const int s   = row % S_SZ;
  const int tok = x[row];
  const float* e = emb + (size_t)tok * D_SZ;
  float* o = h + (size_t)row * D_SZ;
  const float k = -9.210340371976184f / (float)D_SZ;   // -ln(10000)/D
  for (int i = threadIdx.x; i < D_SZ; i += 256) {
    float arg = (float)s * expf((float)(i & ~1) * k);
    float pe  = (i & 1) ? cosf(arg) : sinf(arg);
    o[i] = e[i] * 32.0f + pe;                 // sqrt(1024) = 32
  }
}

// ---------------- LayerNorm: f32 in -> bf16 out ----------------
__global__ __launch_bounds__(256) void ln_bf16_kernel(
    const float* __restrict__ x, const float* __restrict__ w,
    const float* __restrict__ b, unsigned short* __restrict__ out) {
  const int row = blockIdx.x;
  const float* xr = x + (size_t)row * D_SZ;
  __shared__ float s1[256], s2[256];
  float a = 0.f, c = 0.f;
  for (int i = threadIdx.x; i < D_SZ; i += 256) {
    float v = xr[i]; a += v; c += v * v;
  }
  s1[threadIdx.x] = a; s2[threadIdx.x] = c;
  __syncthreads();
  for (int st = 128; st > 0; st >>= 1) {
    if (threadIdx.x < st) {
      s1[threadIdx.x] += s1[threadIdx.x + st];
      s2[threadIdx.x] += s2[threadIdx.x + st];
    }
    __syncthreads();
  }
  const float mean = s1[0] * (1.0f / D_SZ);
  const float var  = s2[0] * (1.0f / D_SZ) - mean * mean;
  const float inv  = rsqrtf(var + 1e-5f);
  for (int i = threadIdx.x; i < D_SZ; i += 256) {
    float v = (xr[i] - mean) * inv * w[i] + b[i];
    out[(size_t)row * D_SZ + i] = f32_to_bf16(v);
  }
}

// ---------------- bf16 WMMA GEMM: C[m,n] = sum_k A[m,k]*W[n,k] ----------------
// Block tile: 128(M) x 128(N), K-step 32, double-buffered async->LDS staging.
// 8 waves in a 4(M) x 2(N) grid; each wave owns a 32x64 strip: 2 A-frags x
// 4 B-frags -> 8 wmma per K-step.
#define BM 128
#define BN 128
#define BK 32
__global__ __launch_bounds__(256) void gemm_bf16_kernel(
    const unsigned short* __restrict__ A,   // (M,K) bf16 row-major
    const unsigned short* __restrict__ W,   // (N,K) bf16 row-major
    const float* bias,                      // (N) or null
    const float* resid,                     // (M,N) or null (may alias outF)
    float* outF,                            // (M,N) or null
    unsigned short* outB,                   // (M,N) or null
    int N, int K, int fuseGelu) {
  __shared__ unsigned short As[2][BM * BK];   // 2 x 8 KB
  __shared__ unsigned short Bs[2][BN * BK];   // 2 x 8 KB
  const int n0 = blockIdx.x * BN;
  const int m0 = blockIdx.y * BM;
  const int tid  = threadIdx.x;
  const int lane = tid & 31, wv = tid >> 5;
  const int wm = wv & 3, wn = wv >> 2;        // wave position in 4x2 grid
  const int kg = lane >> 4, ln = lane & 15;

  const v8f vzero = {0.f,0.f,0.f,0.f,0.f,0.f,0.f,0.f};
  v8f acc[2][4];
#pragma unroll
  for (int mi = 0; mi < 2; ++mi)
#pragma unroll
    for (int bn = 0; bn < 4; ++bn) acc[mi][bn] = vzero;

  const int nT = K / BK;

  // async stage of tile kt into buffer bi (4 x b128 per thread)
  auto stage = [&](int kt, int bi) {
    const int k0 = kt * BK;
#pragma unroll
    for (int i = 0; i < 2; ++i) {
      int c = tid + 256 * i;                 // 0..511
      int r = c >> 2;
      int kc = (c & 3) << 3;
      async_ld_b128(&As[bi][r * BK + kc], &A[(size_t)(m0 + r) * K + k0 + kc]);
    }
#pragma unroll
    for (int i = 0; i < 2; ++i) {
      int c = tid + 256 * i;
      int r = c >> 2;
      int kc = (c & 3) << 3;
      async_ld_b128(&Bs[bi][r * BK + kc], &W[(size_t)(n0 + r) * K + k0 + kc]);
    }
  };

  stage(0, 0);
  for (int kt = 0; kt < nT; ++kt) {
    wait_async0();          // this wave's async writes to buf[kt&1] landed
    __syncthreads();        // all waves' writes visible; prior reads retired
    if (kt + 1 < nT) stage(kt + 1, (kt + 1) & 1);

    const unsigned short* as = As[kt & 1];
    const unsigned short* bs = Bs[kt & 1];
    // A fragments 16x32: lane row = ln, K chunks at 8*kg and 8*kg+16
    FragBF af[2];
#pragma unroll
    for (int mi = 0; mi < 2; ++mi) {
      const unsigned short* ap = &as[(32 * wm + 16 * mi + ln) * BK + 8 * kg];
      af[mi].h[0] = *(const v8bf*)ap;
      af[mi].h[1] = *(const v8bf*)(ap + 16);
    }
#pragma unroll
    for (int bn = 0; bn < 4; ++bn) {
      // B fragment 32x16: lane col = ln, K chunks at 8*kg and 8*kg+16
      FragBF bfr;
      const unsigned short* bp = &bs[(64 * wn + 16 * bn + ln) * BK + 8 * kg];
      bfr.h[0] = *(const v8bf*)bp;
      bfr.h[1] = *(const v8bf*)(bp + 16);
#pragma unroll
      for (int mi = 0; mi < 2; ++mi)
        acc[mi][bn] = __builtin_amdgcn_wmma_f32_16x16x32_bf16(
            false, af[mi].v, false, bfr.v, (short)0, acc[mi][bn], false, false);
    }
  }

  // epilogue: D layout -> lane col = ln, VGPR r -> row r + 8*kg
#pragma unroll
  for (int mi = 0; mi < 2; ++mi)
#pragma unroll
    for (int bn = 0; bn < 4; ++bn) {
      const int col = n0 + 64 * wn + 16 * bn + ln;
      const float bv = bias ? bias[col] : 0.0f;
#pragma unroll
      for (int r = 0; r < 8; ++r) {
        const int row = m0 + 32 * wm + 16 * mi + 8 * kg + r;
        float v = acc[mi][bn][r] + bv;
        if (fuseGelu) v = 0.5f * v * (1.0f + erff(v * 0.70710678118654752f));
        if (resid) v += resid[(size_t)row * N + col];
        if (outF) outF[(size_t)row * N + col] = v;
        if (outB) outB[(size_t)row * N + col] = f32_to_bf16(v);
      }
    }
}

// ---------------- WMMA flash attention (causal) ----------------
// grid: (B*NH, S/64); block: 128 (4 waves); each wave does 16 query rows.
__global__ __launch_bounds__(128) void attn_kernel(
    const unsigned short* __restrict__ qb,
    const unsigned short* __restrict__ kb,
    const unsigned short* __restrict__ vb,
    unsigned short* __restrict__ ctx) {
  __shared__ unsigned short Vs[4][32][64];   // per-wave V stage (16 KB)
  __shared__ unsigned short Ps[4][16][32];   // per-wave P transpose (4 KB)
  const int w    = threadIdx.x >> 5;
  const int lane = threadIdx.x & 31;
  const int kg = lane >> 4, ln = lane & 15;
  const int bh = blockIdx.x;
  const int b  = bh / NH, hd = bh % NH;
  const int q0 = blockIdx.y * 64 + w * 16;

  const unsigned short* qp = qb + (size_t)b * S_SZ * D_SZ + hd * DK;
  const unsigned short* kp = kb + (size_t)b * S_SZ * D_SZ + hd * DK;
  const unsigned short* vp = vb + (size_t)b * S_SZ * D_SZ + hd * DK;
  unsigned short*       cp = ctx + (size_t)b * S_SZ * D_SZ + hd * DK;

  // Q fragments: two 16x32 A-frags covering d=0..31 and 32..63
  FragBF aq[2];
#pragma unroll
  for (int f = 0; f < 2; ++f) {
    const unsigned short* p = qp + (size_t)(q0 + ln) * D_SZ + 32 * f + 8 * kg;
    aq[f].h[0] = *(const v8bf*)p;
    aq[f].h[1] = *(const v8bf*)(p + 16);
  }

  const v8f vzero = {0.f,0.f,0.f,0.f,0.f,0.f,0.f,0.f};
  v8f accO[4] = {vzero, vzero, vzero, vzero};
  float mrow[8], lrow[8];
#pragma unroll
  for (int r = 0; r < 8; ++r) { mrow[r] = -1e30f; lrow[r] = 0.f; }
  const float scale = 0.125f;              // 1/sqrt(64)

  for (int j0 = 0; j0 <= q0 + 15; j0 += 32) {
    // async-stage V block 32x64 into this wave's LDS region
#pragma unroll
    for (int i = 0; i < 8; ++i) {
      int c = lane + 32 * i;
      int r = c >> 3;
      int dc = (c & 7) << 3;
      async_ld_b128(&Vs[w][r][dc], &vp[(size_t)(j0 + r) * D_SZ + dc]);
    }
    // scores: two 16x16 tiles (keys j0..j0+15, j0+16..j0+31)
    v8f sc[2];
#pragma unroll
    for (int t = 0; t < 2; ++t) {
      v8f c = vzero;
#pragma unroll
      for (int f = 0; f < 2; ++f) {
        FragBF bk;   // B 32x16: lane col = key ln, K = d chunks at 32*f + 8*kg and +16
        const unsigned short* p =
            kp + (size_t)(j0 + 16 * t + ln) * D_SZ + 32 * f + 8 * kg;
        bk.h[0] = *(const v8bf*)p;
        bk.h[1] = *(const v8bf*)(p + 16);
        c = __builtin_amdgcn_wmma_f32_16x16x32_bf16(
            false, aq[f].v, false, bk.v, (short)0, c, false, false);
      }
      sc[t] = c;
    }
    // scale + causal mask + row max (row = query)
    float mloc[8];
#pragma unroll
    for (int r = 0; r < 8; ++r) {
      const int qrow = q0 + 8 * kg + r;
      const int key0 = j0 + ln, key1 = j0 + 16 + ln;
      float v0 = (key0 <= qrow) ? sc[0][r] * scale : -1e30f;
      float v1 = (key1 <= qrow) ? sc[1][r] * scale : -1e30f;
      sc[0][r] = v0; sc[1][r] = v1;
      mloc[r] = fmaxf(v0, v1);
    }
#pragma unroll
    for (int msk = 1; msk < 16; msk <<= 1)
#pragma unroll
      for (int r = 0; r < 8; ++r)
        mloc[r] = fmaxf(mloc[r], __shfl_xor(mloc[r], msk, 32));
    // online softmax update; write P (bf16) transposed through LDS
    float corr[8], sloc[8];
#pragma unroll
    for (int r = 0; r < 8; ++r) {
      float mn = fmaxf(mrow[r], mloc[r]);
      corr[r] = expf(mrow[r] - mn);
      mrow[r] = mn;
      float p0 = expf(sc[0][r] - mn);
      float p1 = expf(sc[1][r] - mn);
      sloc[r] = p0 + p1;
      Ps[w][8 * kg + r][ln]      = f32_to_bf16(p0);
      Ps[w][8 * kg + r][16 + ln] = f32_to_bf16(p1);
    }
#pragma unroll
    for (int msk = 1; msk < 16; msk <<= 1)
#pragma unroll
      for (int r = 0; r < 8; ++r)
        sloc[r] += __shfl_xor(sloc[r], msk, 32);
#pragma unroll
    for (int r = 0; r < 8; ++r) lrow[r] = lrow[r] * corr[r] + sloc[r];
#pragma unroll
    for (int sl = 0; sl < 4; ++sl)
#pragma unroll
      for (int r = 0; r < 8; ++r) accO[sl][r] *= corr[r];
    // P A-fragment 16x32 (row = query ln, K = key within block)
    FragBF ap;
    const unsigned short* pp = &Ps[w][ln][8 * kg];
    ap.h[0] = *(const v8bf*)pp;
    ap.h[1] = *(const v8bf*)(pp + 16);
    // V writes must have landed before the transposed gathers below
    wait_async0();
    // P @ V: 4 d-slices of 16
#pragma unroll
    for (int sl = 0; sl < 4; ++sl) {
      FragBF bv;   // B 32x16: lane col = d = 16*sl+ln, K = key = 8*kg + (e&7) + 2*(e&8)
#pragma unroll
      for (int e = 0; e < 16; ++e)
        bv.v[e] = us_to_bf(Vs[w][8 * kg + (e & 7) + ((e & 8) << 1)][16 * sl + ln]);
      accO[sl] = __builtin_amdgcn_wmma_f32_16x16x32_bf16(
          false, ap.v, false, bv.v, (short)0, accO[sl], false, false);
    }
  }
  // normalize and write ctx (bf16)
#pragma unroll
  for (int sl = 0; sl < 4; ++sl)
#pragma unroll
    for (int r = 0; r < 8; ++r) {
      const int qrow = q0 + 8 * kg + r;
      float v = accO[sl][r] / lrow[r];
      cp[(size_t)qrow * D_SZ + 16 * sl + ln] = f32_to_bf16(v);
    }
}

// ---------------- host-side orchestration ----------------
extern "C" void kernel_launch(void* const* d_in, const int* in_sizes, int n_in,
                              void* d_out, int out_size, void* d_ws, size_t ws_size,
                              hipStream_t stream) {
  (void)in_sizes; (void)n_in; (void)out_size; (void)ws_size;
  const int*   x     = (const int*)  d_in[0];
  const float* emb   = (const float*)d_in[1];
  const float* ln1_w = (const float*)d_in[2];
  const float* ln1_b = (const float*)d_in[3];
  const float* wq_w  = (const float*)d_in[4];
  const float* wq_b  = (const float*)d_in[5];
  const float* wk_w  = (const float*)d_in[6];
  const float* wk_b  = (const float*)d_in[7];
  const float* wv_w  = (const float*)d_in[8];
  const float* wv_b  = (const float*)d_in[9];
  const float* wo_w  = (const float*)d_in[10];
  const float* wo_b  = (const float*)d_in[11];
  const float* ln2_w = (const float*)d_in[12];
  const float* ln2_b = (const float*)d_in[13];
  const float* w1_w  = (const float*)d_in[14];
  const float* w1_b  = (const float*)d_in[15];
  const float* w2_w  = (const float*)d_in[16];
  const float* w2_b  = (const float*)d_in[17];
  const float* lnf_w = (const float*)d_in[18];
  const float* lnf_b = (const float*)d_in[19];
  float* out = (float*)d_out;

  // workspace bump allocator (256B aligned)
  char* wsp = (char*)d_ws;
  size_t off = 0;
  auto alloc = [&](size_t bytes) -> void* {
    off = (off + 255) & ~(size_t)255;
    void* p = wsp + off;
    off += bytes;
    return p;
  };
  unsigned short* emb_bf = (unsigned short*)alloc((size_t)V_SZ * D_SZ * 2);
  unsigned short* wq_bf  = (unsigned short*)alloc((size_t)L_NUM * D_SZ * D_SZ * 2);
  unsigned short* wk_bf  = (unsigned short*)alloc((size_t)L_NUM * D_SZ * D_SZ * 2);
  unsigned short* wv_bf  = (unsigned short*)alloc((size_t)L_NUM * D_SZ * D_SZ * 2);
  unsigned short* wo_bf  = (unsigned short*)alloc((size_t)L_NUM * D_SZ * D_SZ * 2);
  unsigned short* w1_bf  = (unsigned short*)alloc((size_t)L_NUM * FF_SZ * D_SZ * 2);
  unsigned short* w2_bf  = (unsigned short*)alloc((size_t)L_NUM * D_SZ * FF_SZ * 2);
  float*          h      = (float*)         alloc((size_t)M_TOK * D_SZ * 4);
  unsigned short* xn_bf  = (unsigned short*)alloc((size_t)M_TOK * D_SZ * 2);
  unsigned short* q_bf   = (unsigned short*)alloc((size_t)M_TOK * D_SZ * 2);
  unsigned short* k_bf   = (unsigned short*)alloc((size_t)M_TOK * D_SZ * 2);
  unsigned short* v_bf   = (unsigned short*)alloc((size_t)M_TOK * D_SZ * 2);
  unsigned short* ctx_bf = (unsigned short*)alloc((size_t)M_TOK * D_SZ * 2);
  unsigned short* ff1_bf = (unsigned short*)alloc((size_t)M_TOK * FF_SZ * 2);
  unsigned short* hn_bf  = (unsigned short*)alloc((size_t)M_TOK * D_SZ * 2);

  auto conv = [&](const float* src, unsigned short* dst, size_t n) {
    cvt_bf16_kernel<<<dim3((unsigned)((n + 255) / 256)), 256, 0, stream>>>(src, dst, n);
  };
  conv(emb,  emb_bf, (size_t)V_SZ * D_SZ);
  conv(wq_w, wq_bf,  (size_t)L_NUM * D_SZ * D_SZ);
  conv(wk_w, wk_bf,  (size_t)L_NUM * D_SZ * D_SZ);
  conv(wv_w, wv_bf,  (size_t)L_NUM * D_SZ * D_SZ);
  conv(wo_w, wo_bf,  (size_t)L_NUM * D_SZ * D_SZ);
  conv(w1_w, w1_bf,  (size_t)L_NUM * FF_SZ * D_SZ);
  conv(w2_w, w2_bf,  (size_t)L_NUM * D_SZ * FF_SZ);

  embed_kernel<<<dim3(M_TOK), 256, 0, stream>>>(x, emb, h);

  auto gemm = [&](const unsigned short* A, const unsigned short* W,
                  const float* bias, const float* resid,
                  float* outF, unsigned short* outB, int N, int K, int gelu) {
    gemm_bf16_kernel<<<dim3(N / BN, M_TOK / BM), 256, 0, stream>>>(
        A, W, bias, resid, outF, outB, N, K, gelu);
  };

  for (int l = 0; l < L_NUM; ++l) {
    const size_t wDD = (size_t)l * D_SZ * D_SZ;
    const size_t wFD = (size_t)l * FF_SZ * D_SZ;
    ln_bf16_kernel<<<dim3(M_TOK), 256, 0, stream>>>(
        h, ln1_w + (size_t)l * D_SZ, ln1_b + (size_t)l * D_SZ, xn_bf);
    gemm(xn_bf, wq_bf + wDD, wq_b + (size_t)l * D_SZ, nullptr, nullptr, q_bf, D_SZ, D_SZ, 0);
    gemm(xn_bf, wk_bf + wDD, wk_b + (size_t)l * D_SZ, nullptr, nullptr, k_bf, D_SZ, D_SZ, 0);
    gemm(xn_bf, wv_bf + wDD, wv_b + (size_t)l * D_SZ, nullptr, nullptr, v_bf, D_SZ, D_SZ, 0);
    attn_kernel<<<dim3(B_SZ * NH, S_SZ / 64), 128, 0, stream>>>(q_bf, k_bf, v_bf, ctx_bf);
    gemm(ctx_bf, wo_bf + wDD, wo_b + (size_t)l * D_SZ, h, h, nullptr, D_SZ, D_SZ, 0);
    ln_bf16_kernel<<<dim3(M_TOK), 256, 0, stream>>>(
        h, ln2_w + (size_t)l * D_SZ, ln2_b + (size_t)l * D_SZ, xn_bf);
    gemm(xn_bf, w1_bf + wFD, w1_b + (size_t)l * FF_SZ, nullptr, nullptr, ff1_bf, FF_SZ, D_SZ, 1);
    gemm(ff1_bf, w2_bf + wFD, w2_b + (size_t)l * D_SZ, h, h, nullptr, D_SZ, FF_SZ, 0);
  }

  ln_bf16_kernel<<<dim3(M_TOK), 256, 0, stream>>>(h, lnf_w, lnf_b, hn_bf);
  // tied output projection: logits = hn @ emb^T  (N = 32000, K = 1024)
  gemm(hn_bf, emb_bf, nullptr, nullptr, out, nullptr, V_SZ, D_SZ, 0);
}
